// Model_5480378270298
// MI455X (gfx1250) — compile-verified, matrix operand order checked by reading, not run on hardware
//
#include <hip/hip_runtime.h>
#include <math.h>
#include <stdint.h>

// ---------------- CDNA5 WMMA plumbing (gfx1250, wave32) ----------------
typedef __bf16 bf16;
typedef __attribute__((ext_vector_type(16))) __bf16 v16bf;
typedef __attribute__((ext_vector_type(8)))  __bf16 v8bf;
typedef __attribute__((ext_vector_type(8)))  float  v8f;

// Sizes
#define SS 48
#define BB 64
#define HE 300
#define HD 256
#define EE 50
#define AA 128
#define TT 50
#define NN 3072     // S*B
#define GN 912      // 3*HE padded to x16
#define KD 320      // 300 padded to x32
#define K2 608      // 600 padded to x32
#define GD 768      // 3*HD
#define KE 64       // 50 padded to x32
#define KH 256

// 16x32 bf16 fragment load (ISA 7.12.2 striping):
// lanes 0-15 : maj=lane,    K {0..7} then {16..23}
// lanes 16-31: maj=lane-16, K {8..15} then {24..31}
__device__ inline v16bf load_frag(const bf16* __restrict__ base, int ld) {
  const int lane = threadIdx.x & 31;
  const int maj  = lane & 15;
  const int kb   = (lane & 16) ? 8 : 0;
  const bf16* p = base + (long)maj * ld;
  v8bf lo = *(const v8bf*)(p + kb);
  v8bf hi = *(const v8bf*)(p + kb + 16);
  v16bf f;
#pragma unroll
  for (int i = 0; i < 8; ++i) { f[i] = lo[i]; f[i + 8] = hi[i]; }
  return f;
}

__device__ inline void wmma_step(v8f& acc, v16bf a, v16bf b) {
  acc = __builtin_amdgcn_wmma_f32_16x16x32_bf16(false, a, false, b, (short)0, acc,
                                                false, false);
}

// C/D 16x16 f32 layout: lanes 0-15 -> N=lane, M=v; lanes 16-31 -> N=lane-16, M=v+8
__device__ inline void store_tile(float* __restrict__ C, int ldc, v8f acc,
                                  int m0, int n0, int nlimit,
                                  const float* __restrict__ bias, int act) {
  const int lane = threadIdx.x & 31;
  const int col  = n0 + (lane & 15);
  const int mofs = (lane & 16) ? 8 : 0;
  if (col >= nlimit) return;
  const float bv = bias ? bias[col] : 0.0f;
#pragma unroll
  for (int v = 0; v < 8; ++v) {
    float x = acc[v] + bv;
    if (act == 1) x = tanhf(x);
    C[(long)(m0 + mofs + v) * ldc + col] = x;
  }
}

// Single 16x16 tile accumulate over K
__device__ inline void wave_gemm_acc(v8f& acc, const bf16* __restrict__ A, int lda,
                                     const bf16* __restrict__ W, int ldb, int kp) {
  for (int k = 0; k < kp; k += 32) {
    if (k + 32 < kp) __builtin_prefetch(W + k + 256, 0, 1);   // global_prefetch_b8
    v16bf a = load_frag(A + k, lda);
    v16bf b = load_frag(W + k, ldb);
    wmma_step(acc, a, b);
  }
}

// 16x64 output strip: one A fragment reused across 4 B tiles (4 accumulators)
__device__ inline void wave_gemm_strip4(v8f acc[4], const bf16* __restrict__ A, int lda,
                                        const bf16* __restrict__ W, int ldb, int kp) {
  for (int k = 0; k < kp; k += 32) {
    if (k + 32 < kp) __builtin_prefetch(A + k + 32, 0, 1);
    v16bf a = load_frag(A + k, lda);
#pragma unroll
    for (int j = 0; j < 4; ++j) {
      v16bf b = load_frag(W + (long)(j * 16) * ldb + k, ldb);
      wmma_step(acc[j], a, b);
    }
  }
}

__device__ inline float sigf(float x) { return 1.0f / (1.0f + expf(-x)); }

// Async copy 16B global -> LDS (ASYNCcnt path, ISA 15.18.3 op 98)
__device__ inline void async_g2l_b128(uint32_t lds_addr, const bf16* gsrc) {
  asm volatile("global_load_async_to_lds_b128 %0, %1, off"
               :: "v"(lds_addr), "v"(gsrc) : "memory");
}
__device__ inline void wait_async0() {
#if __has_builtin(__builtin_amdgcn_s_wait_asynccnt)
  __builtin_amdgcn_s_wait_asynccnt(0);
#else
  asm volatile("s_wait_asynccnt 0x0" ::: "memory");
#endif
}

// ---------------- generic kernels ----------------
__global__ void k_cvt_pad(const float* __restrict__ src, int R, int C,
                          bf16* __restrict__ dst, int Rp, int Cp) {
  long n = (long)Rp * Cp;
  for (long i = blockIdx.x * (long)blockDim.x + threadIdx.x; i < n;
       i += (long)gridDim.x * blockDim.x) {
    int r = (int)(i / Cp), c = (int)(i % Cp);
    dst[i] = (r < R && c < C) ? (bf16)src[(long)r * C + c] : (bf16)0.0f;
  }
}

__global__ void k_add_cvt_pad(const float* __restrict__ a, const float* __restrict__ b,
                              int R, int C, bf16* __restrict__ dst, int Cp) {
  long n = (long)R * Cp;
  for (long i = blockIdx.x * (long)blockDim.x + threadIdx.x; i < n;
       i += (long)gridDim.x * blockDim.x) {
    int r = (int)(i / Cp), c = (int)(i % Cp);
    dst[i] = (c < C) ? (bf16)(a[(long)r * C + c] + b[(long)r * C + c]) : (bf16)0.0f;
  }
}

// C(M x Nlog) = A(M x kp) * W(Ntiles*16 x kp)^T + bias, optional tanh
__global__ __launch_bounds__(256)
void k_gemm(const bf16* __restrict__ A, int lda, const bf16* __restrict__ W, int ldb,
            const float* __restrict__ bias, float* __restrict__ C, int ldc,
            int mtiles, int ntiles, int nlimit, int kp, int act) {
  const int wave   = (blockIdx.x * blockDim.x + threadIdx.x) >> 5;
  const int nwaves = (gridDim.x * blockDim.x) >> 5;
  const int tiles  = mtiles * ntiles;
  for (int t = wave; t < tiles; t += nwaves) {
    const int mt = t / ntiles, nt = t % ntiles;
    v8f acc = {};
    wave_gemm_acc(acc, A + (long)mt * 16 * lda, lda, W + (long)nt * 16 * ldb, ldb, kp);
    store_tile(C, ldc, acc, mt * 16, nt * 16, nlimit, bias, act);
  }
}

// ---------------- encoder recurrent (one workgroup per direction) ----------------
struct EncRec {
  const float* gi;   // (S*64, 900) precomputed x@Wih^T + bih
  const bf16*  Whh;  // (912, 320) padded bf16
  const float* bhh;  // (900)
  float* h;          // (64,300) fp32 state
  bf16*  hb;         // (64,320) bf16 state (zero padded)
  float* gh;         // (64,912) f32 scratch
  float* ys;         // output base (row-major, ld = ysLd)
  int ysLd, ysOff, rev;
};

__global__ __launch_bounds__(256)
void k_enc_rec(EncRec p0, EncRec p1) {
  EncRec p = (blockIdx.x == 0) ? p0 : p1;
  const int tid = threadIdx.x, wave = tid >> 5;
  for (int i = tid; i < 64 * 300; i += 256) p.h[i] = 0.0f;
  for (int i = tid; i < 64 * KD; i += 256) p.hb[i] = (bf16)0.0f;
  __threadfence(); __syncthreads();
  for (int s = 0; s < SS; ++s) {
    const int t = p.rev ? (SS - 1 - s) : s;
    // gh = h @ Whh^T : N=912 -> 14 strips of 64 + 1 tile, x 4 m-tiles
    for (int jb = wave; jb < 60; jb += 8) {
      if (jb < 56) {
        const int mt = jb / 14, ns = jb % 14;
        v8f acc[4] = {};
        wave_gemm_strip4(acc, p.hb + mt * 16 * KD, KD,
                         p.Whh + (long)ns * 64 * KD, KD, KD);
#pragma unroll
        for (int j = 0; j < 4; ++j)
          store_tile(p.gh, GN, acc[j], mt * 16, ns * 64 + j * 16, GN, nullptr, 0);
      } else {
        const int mt = jb - 56;
        v8f acc = {};
        wave_gemm_acc(acc, p.hb + mt * 16 * KD, KD, p.Whh + (long)56 * 16 * KD, KD, KD);
        store_tile(p.gh, GN, acc, mt * 16, 56 * 16, GN, nullptr, 0);
      }
    }
    __threadfence(); __syncthreads();
    const float* gib = p.gi + (long)(t * BB) * 900;
    for (int i = tid; i < 64 * 300; i += 256) {
      const int r = i / 300, j = i % 300;
      const float* gir = gib + (long)r * 900;
      const float ghr = p.gh[r * GN + j]       + p.bhh[j];
      const float ghz = p.gh[r * GN + 300 + j] + p.bhh[300 + j];
      const float ghn = p.gh[r * GN + 600 + j] + p.bhh[600 + j];
      const float hv  = p.h[i];
      const float rg = sigf(gir[j] + ghr);
      const float zg = sigf(gir[300 + j] + ghz);
      const float ng = tanhf(gir[600 + j] + rg * ghn);
      const float hn = (1.0f - zg) * ng + zg * hv;
      p.h[i] = hn;
      p.hb[r * KD + j] = (bf16)hn;
      p.ys[(long)(t * BB + r) * p.ysLd + p.ysOff + j] = hn;
    }
    __threadfence(); __syncthreads();
  }
}

// ---------------- decoder: persistent, 64 rows per workgroup ----------------
struct DecArgs {
  const bf16 *Wih0, *Whh0, *Wih1, *Whh1, *Wo;   // padded bf16
  const float *bih0, *bhh0, *bih1, *bhh1, *bo;
  const bf16* embp;    // (128, 64) bf16, zero padded cols 50..63
  float* H;            // (3072, 512) [h0|h1] fp32, updated in place
  bf16 *hb0, *hb1;     // per-wg (64, 256) bf16 staging
  float *Gi, *Gh;      // per-wg (64, 768) gate scratch
  float* out;          // (49, 3072, 128)
};

__global__ __launch_bounds__(256)
void k_decoder(DecArgs a) {
  const int wg = blockIdx.x, tid = threadIdx.x, wave = tid >> 5;
  const int row0 = wg * 64;
  float* H   = a.H   + (long)row0 * 512;
  bf16*  hb0 = a.hb0 + (long)wg * 64 * KH;
  bf16*  hb1 = a.hb1 + (long)wg * 64 * KH;
  float* Gi  = a.Gi  + (long)wg * 64 * GD;
  float* Gh  = a.Gh  + (long)wg * 64 * GD;

  __shared__ __align__(16) bf16  eb[64 * KE];
  __shared__ __align__(16) float logits[64 * AA];
  __shared__ float lse[64];
  __shared__ int   pidx[64];

  if (tid < 64) pidx[tid] = 1;            // SOS token
  for (int i = tid; i < 64 * KH; i += 256) {
    const int r = i >> 8, c = i & 255;
    hb0[i] = (bf16)H[(long)r * 512 + c];
    hb1[i] = (bf16)H[(long)r * 512 + 256 + c];
  }
  __threadfence(); __syncthreads();
  // async-gather eb rows (16B/lane) from embp via ASYNCcnt path
#pragma unroll
  for (int k = 0; k < 2; ++k) {
    const int c = tid + k * 256;
    const int r = c >> 3, sub = c & 7;
    async_g2l_b128((uint32_t)(uintptr_t)(eb + r * KE + sub * 8),
                   a.embp + (long)pidx[r] * KE + sub * 8);
  }
  wait_async0();
  __syncthreads();

  for (int t = 0; t < TT - 1; ++t) {
    // ---- layer 0: Gi = e @ Wih0^T (K=64), Gh = h0 @ Whh0^T (K=256); N=768 = 12 strips
    for (int st = wave; st < 4 * 12; st += 8) {
      const int mt = st / 12, ns = st % 12;
      v8f gi4[4] = {};
      wave_gemm_strip4(gi4, eb + mt * 16 * KE, KE, a.Wih0 + (long)ns * 64 * KE, KE, KE);
      v8f gh4[4] = {};
      wave_gemm_strip4(gh4, hb0 + mt * 16 * KH, KH, a.Whh0 + (long)ns * 64 * KH, KH, KH);
#pragma unroll
      for (int j = 0; j < 4; ++j) {
        store_tile(Gi, GD, gi4[j], mt * 16, ns * 64 + j * 16, GD, nullptr, 0);
        store_tile(Gh, GD, gh4[j], mt * 16, ns * 64 + j * 16, GD, nullptr, 0);
      }
    }
    __threadfence(); __syncthreads();
    for (int i = tid; i < 64 * KH; i += 256) {
      const int r = i >> 8, j = i & 255;
      const float gir = Gi[r * GD + j]       + a.bih0[j];
      const float giz = Gi[r * GD + 256 + j] + a.bih0[256 + j];
      const float gin = Gi[r * GD + 512 + j] + a.bih0[512 + j];
      const float ghr = Gh[r * GD + j]       + a.bhh0[j];
      const float ghz = Gh[r * GD + 256 + j] + a.bhh0[256 + j];
      const float ghn = Gh[r * GD + 512 + j] + a.bhh0[512 + j];
      const float hv  = H[(long)r * 512 + j];
      const float rg = sigf(gir + ghr), zg = sigf(giz + ghz);
      const float ng = tanhf(gin + rg * ghn);
      const float hn = (1.0f - zg) * ng + zg * hv;
      H[(long)r * 512 + j] = hn;
      hb0[i] = (bf16)hn;
    }
    __threadfence(); __syncthreads();
    // ---- layer 1: Gi = h0n @ Wih1^T, Gh = h1 @ Whh1^T (K=256)
    for (int st = wave; st < 4 * 12; st += 8) {
      const int mt = st / 12, ns = st % 12;
      v8f gi4[4] = {};
      wave_gemm_strip4(gi4, hb0 + mt * 16 * KH, KH, a.Wih1 + (long)ns * 64 * KH, KH, KH);
      v8f gh4[4] = {};
      wave_gemm_strip4(gh4, hb1 + mt * 16 * KH, KH, a.Whh1 + (long)ns * 64 * KH, KH, KH);
#pragma unroll
      for (int j = 0; j < 4; ++j) {
        store_tile(Gi, GD, gi4[j], mt * 16, ns * 64 + j * 16, GD, nullptr, 0);
        store_tile(Gh, GD, gh4[j], mt * 16, ns * 64 + j * 16, GD, nullptr, 0);
      }
    }
    __threadfence(); __syncthreads();
    for (int i = tid; i < 64 * KH; i += 256) {
      const int r = i >> 8, j = i & 255;
      const float gir = Gi[r * GD + j]       + a.bih1[j];
      const float giz = Gi[r * GD + 256 + j] + a.bih1[256 + j];
      const float gin = Gi[r * GD + 512 + j] + a.bih1[512 + j];
      const float ghr = Gh[r * GD + j]       + a.bhh1[j];
      const float ghz = Gh[r * GD + 256 + j] + a.bhh1[256 + j];
      const float ghn = Gh[r * GD + 512 + j] + a.bhh1[512 + j];
      const float hv  = H[(long)r * 512 + 256 + j];
      const float rg = sigf(gir + ghr), zg = sigf(giz + ghz);
      const float ng = tanhf(gin + rg * ghn);
      const float hn = (1.0f - zg) * ng + zg * hv;
      H[(long)r * 512 + 256 + j] = hn;
      hb1[i] = (bf16)hn;
    }
    __threadfence(); __syncthreads();
    // ---- logits = h1n @ Wout^T + bout (into LDS): 4 m-tiles x 2 strips
    for (int st = wave; st < 8; st += 8) {
      const int mt = st >> 1, ns = st & 1;
      v8f acc[4] = {};
      wave_gemm_strip4(acc, hb1 + mt * 16 * KH, KH, a.Wo + (long)ns * 64 * KH, KH, KH);
#pragma unroll
      for (int j = 0; j < 4; ++j)
        store_tile(logits, AA, acc[j], mt * 16, ns * 64 + j * 16, AA, a.bo, 0);
    }
    __syncthreads();
    if (tid < 64) {
      const float* L = logits + tid * AA;
      float m = L[0]; int pi = 0;
      for (int c = 1; c < AA; ++c) { const float v = L[c]; if (v > m) { m = v; pi = c; } }
      float s = 0.0f;
      for (int c = 0; c < AA; ++c) s += expf(L[c] - m);
      lse[tid] = m + logf(s);
      pidx[tid] = pi;
    }
    __syncthreads();
    float* yo = a.out + (long)t * NN * AA + (long)row0 * AA;
    for (int i = tid; i < 64 * AA; i += 256) yo[i] = logits[i] - lse[i >> 7];
    // next-step embeddings: async gather emb[pidx] -> LDS
#pragma unroll
    for (int k = 0; k < 2; ++k) {
      const int c = tid + k * 256;
      const int r = c >> 3, sub = c & 7;
      async_g2l_b128((uint32_t)(uintptr_t)(eb + r * KE + sub * 8),
                     a.embp + (long)pidx[r] * KE + sub * 8);
    }
    wait_async0();
    __threadfence(); __syncthreads();
  }
}

// ---------------- host orchestration ----------------
extern "C" void kernel_launch(void* const* d_in, const int* in_sizes, int n_in,
                              void* d_out, int out_size, void* d_ws, size_t ws_size,
                              hipStream_t stream) {
  const float* wv     = (const float*)d_in[0];
  const float* e0fWih = (const float*)d_in[1];  const float* e0fWhh = (const float*)d_in[2];
  const float* e0fbih = (const float*)d_in[3];  const float* e0fbhh = (const float*)d_in[4];
  const float* e0bWih = (const float*)d_in[5];  const float* e0bWhh = (const float*)d_in[6];
  const float* e0bbih = (const float*)d_in[7];  const float* e0bbhh = (const float*)d_in[8];
  const float* e1fWih = (const float*)d_in[9];  const float* e1fWhh = (const float*)d_in[10];
  const float* e1fbih = (const float*)d_in[11]; const float* e1fbhh = (const float*)d_in[12];
  const float* e1bWih = (const float*)d_in[13]; const float* e1bWhh = (const float*)d_in[14];
  const float* e1bbih = (const float*)d_in[15]; const float* e1bbhh = (const float*)d_in[16];
  const float* d0Wih  = (const float*)d_in[17]; const float* d0Whh  = (const float*)d_in[18];
  const float* d0bih  = (const float*)d_in[19]; const float* d0bhh  = (const float*)d_in[20];
  const float* d1Wih  = (const float*)d_in[21]; const float* d1Whh  = (const float*)d_in[22];
  const float* d1bih  = (const float*)d_in[23]; const float* d1bhh  = (const float*)d_in[24];
  const float* emb    = (const float*)d_in[25];
  const float* Wout   = (const float*)d_in[26]; const float* bout   = (const float*)d_in[27];
  const float* Wh0    = (const float*)d_in[28]; const float* bh0    = (const float*)d_in[29];
  float* out = (float*)d_out;
  (void)in_sizes; (void)n_in; (void)out_size; (void)ws_size;

  size_t off = 0;
  auto alloc = [&](size_t bytes) -> void* {
    off = (off + 255) & ~(size_t)255;
    void* p = (char*)d_ws + off;
    off += bytes;
    return p;
  };
  bf16* e0fWihp = (bf16*)alloc((size_t)GN * KD * 2);
  bf16* e0bWihp = (bf16*)alloc((size_t)GN * KD * 2);
  bf16* e0fWhhp = (bf16*)alloc((size_t)GN * KD * 2);
  bf16* e0bWhhp = (bf16*)alloc((size_t)GN * KD * 2);
  bf16* e1fWihp = (bf16*)alloc((size_t)GN * K2 * 2);
  bf16* e1bWihp = (bf16*)alloc((size_t)GN * K2 * 2);
  bf16* e1fWhhp = (bf16*)alloc((size_t)GN * KD * 2);
  bf16* e1bWhhp = (bf16*)alloc((size_t)GN * KD * 2);
  bf16* Wh0p    = (bf16*)alloc((size_t)512 * KD * 2);
  bf16* d0Wihp  = (bf16*)alloc((size_t)GD * KE * 2);
  bf16* d0Whhp  = (bf16*)alloc((size_t)GD * KH * 2);
  bf16* d1Wihp  = (bf16*)alloc((size_t)GD * KH * 2);
  bf16* d1Whhp  = (bf16*)alloc((size_t)GD * KH * 2);
  bf16* Woutp   = (bf16*)alloc((size_t)AA * KH * 2);
  bf16* embp    = (bf16*)alloc((size_t)AA * KE * 2);
  bf16*  wvb  = (bf16*)alloc((size_t)NN * KD * 2);
  float* Gi0f = (float*)alloc((size_t)NN * 900 * 4);   // reused for layer-1 gi
  float* Gi0b = (float*)alloc((size_t)NN * 900 * 4);
  float* x0   = (float*)alloc((size_t)NN * 600 * 4);
  bf16*  x0b  = (bf16*)alloc((size_t)NN * K2 * 2);
  float* x1f  = (float*)alloc((size_t)NN * 300 * 4);
  float* x1b  = (float*)alloc((size_t)NN * 300 * 4);
  bf16*  encb = (bf16*)alloc((size_t)NN * KD * 2);
  float* Hbuf = (float*)alloc((size_t)NN * 512 * 4);
  float* hA   = (float*)alloc((size_t)64 * 300 * 4);
  bf16*  hbA  = (bf16*)alloc((size_t)64 * KD * 2);
  float* ghA  = (float*)alloc((size_t)64 * GN * 4);
  float* hB   = (float*)alloc((size_t)64 * 300 * 4);
  bf16*  hbB  = (bf16*)alloc((size_t)64 * KD * 2);
  float* ghB  = (float*)alloc((size_t)64 * GN * 4);
  bf16*  dhb0 = (bf16*)alloc((size_t)48 * 64 * KH * 2);
  bf16*  dhb1 = (bf16*)alloc((size_t)48 * 64 * KH * 2);
  float* dGi  = (float*)alloc((size_t)48 * 64 * GD * 4);
  float* dGh  = (float*)alloc((size_t)48 * 64 * GD * 4);

  auto CVT = [&](const float* s, int R, int C, bf16* d, int Rp, int Cp) {
    long n = (long)Rp * Cp;
    int blocks = (int)((n + 255) / 256); if (blocks > 4096) blocks = 4096;
    k_cvt_pad<<<blocks, 256, 0, stream>>>(s, R, C, d, Rp, Cp);
  };
  CVT(e0fWih, 900, 300, e0fWihp, GN, KD);  CVT(e0bWih, 900, 300, e0bWihp, GN, KD);
  CVT(e0fWhh, 900, 300, e0fWhhp, GN, KD);  CVT(e0bWhh, 900, 300, e0bWhhp, GN, KD);
  CVT(e1fWih, 900, 600, e1fWihp, GN, K2);  CVT(e1bWih, 900, 600, e1bWihp, GN, K2);
  CVT(e1fWhh, 900, 300, e1fWhhp, GN, KD);  CVT(e1bWhh, 900, 300, e1bWhhp, GN, KD);
  CVT(Wh0, 512, 300, Wh0p, 512, KD);
  CVT(d0Wih, GD, 50, d0Wihp, GD, KE);      CVT(d0Whh, GD, KH, d0Whhp, GD, KH);
  CVT(d1Wih, GD, KH, d1Wihp, GD, KH);      CVT(d1Whh, GD, KH, d1Whhp, GD, KH);
  CVT(Wout, AA, KH, Woutp, AA, KH);
  CVT(emb, AA, EE, embp, AA, KE);
  CVT(wv, NN, 300, wvb, NN, KD);

  auto GEMM = [&](const bf16* A, int lda, const bf16* W, int ldb, const float* bias,
                  float* C, int ldc, int M, int ntiles, int nlimit, int kp, int act) {
    int tiles = (M / 16) * ntiles;
    int blocks = (tiles + 7) / 8; if (blocks > 2048) blocks = 2048;
    k_gemm<<<blocks, 256, 0, stream>>>(A, lda, W, ldb, bias, C, ldc,
                                       M / 16, ntiles, nlimit, kp, act);
  };
  // encoder layer-0 input gates
  GEMM(wvb, KD, e0fWihp, KD, e0fbih, Gi0f, 900, NN, 57, 900, KD, 0);
  GEMM(wvb, KD, e0bWihp, KD, e0bbih, Gi0b, 900, NN, 57, 900, KD, 0);
  {
    EncRec pf{Gi0f, e0fWhhp, e0fbhh, hA, hbA, ghA, x0, 600, 0,   0};
    EncRec pb{Gi0b, e0bWhhp, e0bbhh, hB, hbB, ghB, x0, 600, 300, 1};
    k_enc_rec<<<2, 256, 0, stream>>>(pf, pb);
  }
  CVT(x0, NN, 600, x0b, NN, K2);
  // encoder layer-1 (reuse Gi buffers)
  GEMM(x0b, K2, e1fWihp, K2, e1fbih, Gi0f, 900, NN, 57, 900, K2, 0);
  GEMM(x0b, K2, e1bWihp, K2, e1bbih, Gi0b, 900, NN, 57, 900, K2, 0);
  {
    EncRec pf{Gi0f, e1fWhhp, e1fbhh, hA, hbA, ghA, x1f, 300, 0, 0};
    EncRec pb{Gi0b, e1bWhhp, e1bbhh, hB, hbB, ghB, x1b, 300, 0, 1};
    k_enc_rec<<<2, 256, 0, stream>>>(pf, pb);
  }
  {
    long n = (long)NN * KD;
    int blocks = (int)((n + 255) / 256); if (blocks > 4096) blocks = 4096;
    k_add_cvt_pad<<<blocks, 256, 0, stream>>>(x1f, x1b, NN, 300, encb, KD);
  }
  // decoder initial hidden: H = tanh(enc @ Wh0^T + bh0), (3072 x 512)
  GEMM(encb, KD, Wh0p, KD, bh0, Hbuf, 512, NN, 32, 512, KD, 1);
  // persistent greedy decoder
  DecArgs da{d0Wihp, d0Whhp, d1Wihp, d1Whhp, Woutp,
             d0bih, d0bhh, d1bih, d1bhh, bout,
             embp, Hbuf, dhb0, dhb1, dGi, dGh, out};
  k_decoder<<<48, 256, 0, stream>>>(da);
}
